// YuLanMiniAttention_11965778887186
// MI455X (gfx1250) — matrix operand order repaired and checked by reading, not compiled, not run on hardware
//
#include <hip/hip_runtime.h>
#include <hip/hip_bf16.h>

// ---- problem constants (match reference) ----
#define H_    30
#define KVH_  6
#define D_    64
#define HID_  1920
#define S_    2048
#define HD_   (H_*D_)     // 1920
#define KVD_  (KVH_*D_)   // 384
#define WINDOW_ 1024

typedef _Float16 f16;
typedef _Float16 v16h __attribute__((ext_vector_type(16)));
typedef _Float16 v8h  __attribute__((ext_vector_type(8)));
typedef float    v8f  __attribute__((ext_vector_type(8)));
typedef unsigned int u32x4 __attribute__((ext_vector_type(4)));
typedef unsigned int u32x8 __attribute__((ext_vector_type(8)));

static __device__ __forceinline__ v16h cat8(v8h lo, v8h hi) {
  v16h r;
#pragma unroll
  for (int i = 0; i < 8; ++i) { r[i] = lo[i]; r[i + 8] = hi[i]; }
  return r;
}
static __device__ __forceinline__ v8h ld8(const f16* p) { return *(const v8h*)p; }

// ---------------------------------------------------------------------------
// Tensor Data Mover: DMA a 2-D f16 tile [rows x 32] (row stride = K elements)
// from global into LDS, LDS rows padded 64B data + 16B pad (bank-conflict-free).
// Descriptor per CDNA5 ISA; 2-group (<=2D) form.
// ---------------------------------------------------------------------------
#define LDS_ROW 80  // bytes per padded LDS row (64 data + 16 pad)

static __device__ __forceinline__ void tdm_load_tile(
    unsigned int lds_addr, const f16* gsrc, int K_elems, int rows) {
  unsigned long long ga = (unsigned long long)(size_t)gsrc;
  u32x4 g0;
  g0[0] = 1u;                                   // count=1, is_restore=0, gather=0
  g0[1] = lds_addr;                             // lds_addr [63:32]
  g0[2] = (unsigned int)ga;                     // global_addr[31:0]
  g0[3] = ((unsigned int)(ga >> 32) & 0x01FFFFFFu) | (2u << 30);  // addr[56:32] | type=2
  u32x8 g1;
  // data_size=1 (2B), pad_enable=1, pad_interval=3 (64B), pad_amount=3 (16B)
  g1[0] = (1u << 16) | (1u << 20) | (3u << 22) | (3u << 25);
  const unsigned int dim = 1u << 30;            // tensor dims huge: tile always in-bounds
  g1[1] = (dim & 0xFFFFu) << 16;                // tensor_dim0[15:0]
  g1[2] = ((dim >> 16) & 0xFFFFu) | ((dim & 0xFFFFu) << 16);   // dim0[31:16] | dim1[15:0]
  g1[3] = ((dim >> 16) & 0xFFFFu) | (32u << 16);               // dim1[31:16] | tile_dim0=32
  g1[4] = (unsigned int)rows;                   // tile_dim1 = rows, tile_dim2 = 0
  g1[5] = (unsigned int)K_elems;                // tensor_dim0_stride[31:0]
  g1[6] = 0u;                                   // stride0[47:32] | stride1[15:0]
  g1[7] = 0u;
  asm volatile("tensor_load_to_lds %0, %1" :: "s"(g0), "s"(g1) : "memory");
}

// ---------------------------------------------------------------------------
// f32 -> f16 conversion
// ---------------------------------------------------------------------------
__global__ void cvt_f32_to_f16(const float* __restrict__ src,
                               f16* __restrict__ dst, int n) {
  int i = blockIdx.x * blockDim.x + threadIdx.x;
  if (i < n) dst[i] = (f16)src[i];
}

// ---------------------------------------------------------------------------
// WMMA GEMM:  C[M,N] = alpha * A[M,K] @ W[N,K]^T   (f16 in, f32 acc)
// MODE 0: f32 out [M,N];  MODE 2: f16 out [N,M] (transposed)
// B tile (64 cols x 32 k, shared by all 8 waves) staged in LDS by the TDM,
// double-buffered, fenced by s_wait_tensorcnt + barriers.
// A fragments load straight from global (per-lane rows, no reuse across waves).
// ---------------------------------------------------------------------------
template <int MODE>
__global__ __launch_bounds__(256) void gemm_wmma(
    const f16* __restrict__ A, const f16* __restrict__ W,
    void* __restrict__ out, const float* __restrict__ alphaPtr,
    int M, int N, int K) {
  __shared__ __align__(16) unsigned char Bsm[2][64 * LDS_ROW];

  const int lane = threadIdx.x & 31;
  const int wave = threadIdx.x >> 5;
  const int l16  = lane & 15;
  const bool hi  = lane >= 16;
  const int tileN = blockIdx.x * 64;
  const int tileM = blockIdx.y * 128 + wave * 16;

  const f16* Ap = A + (size_t)(tileM + l16) * K + (hi ? 8 : 0);
  const f16* Wt = W + (size_t)tileN * K;   // B tile origin (rows = output cols)
  const int nk = K / 32;

  const unsigned int lds0 = (unsigned int)(size_t)&Bsm[0][0];
  const unsigned int lds1 = (unsigned int)(size_t)&Bsm[1][0];

  v8f acc[4] = {};
  if (wave == 0) tdm_load_tile(lds0, Wt, K, 64);

  for (int j = 0; j < nk; ++j) {
    if (wave == 0) {
      if (j + 1 < nk) {
        tdm_load_tile((j & 1) ? lds0 : lds1, Wt + (j + 1) * 32, K, 64);
        __builtin_amdgcn_s_wait_tensorcnt((short)1);  // tile j landed (in-order)
      } else {
        __builtin_amdgcn_s_wait_tensorcnt((short)0);
      }
    }
    __syncthreads();  // tile j visible to all waves

    const unsigned char* bbuf = Bsm[j & 1];
    v16h a = cat8(ld8(Ap + j * 32), ld8(Ap + j * 32 + 16));
#pragma unroll
    for (int nt = 0; nt < 4; ++nt) {
      const unsigned char* bp = bbuf + (size_t)(nt * 16 + l16) * LDS_ROW + (hi ? 16 : 0);
      v16h b = cat8(*(const v8h*)bp, *(const v8h*)(bp + 32));
      acc[nt] = __builtin_amdgcn_wmma_f32_16x16x32_f16(
          false, a, false, b, (short)0, acc[nt], false, false);
    }
    __syncthreads();  // tile j consumed; its buffer may be re-filled
  }

  const float alpha = alphaPtr[0];
#pragma unroll
  for (int nt = 0; nt < 4; ++nt) {
    const int col = tileN + nt * 16 + l16;
#pragma unroll
    for (int r = 0; r < 8; ++r) {
      const int m = tileM + r + (hi ? 8 : 0);
      const float v = acc[nt][r] * alpha;
      if (MODE == 0)      ((float*)out)[(size_t)m * N + col] = v;
      else                ((f16*)out)[(size_t)col * M + m] = (f16)v;  // transposed
    }
  }
}

// ---------------------------------------------------------------------------
// Per-head LayerNorm (biased var, eps=1e-5) + RoPE, f32 in -> f16 out.
// One wave32 per (s, head); 2 elements per lane: d=lane and d=lane+32.
// ---------------------------------------------------------------------------
__global__ __launch_bounds__(128) void ln_rope_kernel(
    const float* __restrict__ Qf, const float* __restrict__ Kf,
    const float* __restrict__ qlnw, const float* __restrict__ klnw,
    const long long* __restrict__ pos,
    f16* __restrict__ Qh, f16* __restrict__ Kh) {
  const int gw   = blockIdx.x * 4 + (threadIdx.x >> 5);
  const int lane = threadIdx.x & 31;
  const int s  = gw / (H_ + KVH_);
  const int hh = gw % (H_ + KVH_);

  const float* src; const float* w; f16* dst;
  if (hh < H_) {
    src = Qf + (size_t)s * HD_ + hh * D_;
    w   = qlnw + hh * D_;
    dst = Qh + (size_t)s * HD_ + hh * D_;
  } else {
    const int kv = hh - H_;
    src = Kf + (size_t)s * KVD_ + kv * D_;
    w   = klnw + kv * D_;
    dst = Kh + (size_t)s * KVD_ + kv * D_;
  }

  float x0 = src[lane], x1 = src[lane + 32];
  float sum = x0 + x1;
#pragma unroll
  for (int o = 16; o >= 1; o >>= 1) sum += __shfl_xor(sum, o, 32);
  const float mu = sum * (1.f / 64.f);
  const float d0 = x0 - mu, d1 = x1 - mu;
  float vs = d0 * d0 + d1 * d1;
#pragma unroll
  for (int o = 16; o >= 1; o >>= 1) vs += __shfl_xor(vs, o, 32);
  const float rs = rsqrtf(vs * (1.f / 64.f) + 1e-5f);
  const float y0 = d0 * rs * w[lane];
  const float y1 = d1 * rs * w[lane + 32];

  const float p = (float)pos[s];
  const float invf = __expf(-((float)lane * (1.f / 32.f)) * 9.210340371976184f);
  const float ang = p * invf;
  const float c = __cosf(ang), sn = __sinf(ang);
  dst[lane]      = (f16)(y0 * c - y1 * sn);
  dst[lane + 32] = (f16)(y1 * c + y0 * sn);
}

// ---------------------------------------------------------------------------
// Flash attention with sliding-window causal mask (unchanged from round 1).
// ---------------------------------------------------------------------------
__global__ __launch_bounds__(128) void attn_kernel(
    const f16* __restrict__ Qh, const f16* __restrict__ Kh,
    const f16* __restrict__ Vt, f16* __restrict__ Out) {
  const int lane = threadIdx.x & 31;
  const int wave = threadIdx.x >> 5;
  const int head = blockIdx.x >> 5;   // / (S_/64)
  const int qt   = blockIdx.x & 31;
  const int l16  = lane & 15;
  const bool hi  = lane >= 16;
  const int qb   = qt * 64 + wave * 16;
  const int q    = qb + l16;
  const int kvh  = head / (H_ / KVH_);

  const f16* qp = Qh + (size_t)q * HD_ + head * D_ + (hi ? 8 : 0);
  const v16h bq0 = cat8(ld8(qp), ld8(qp + 16));       // d = 0..31
  const v16h bq1 = cat8(ld8(qp + 32), ld8(qp + 48));  // d = 32..63

  v8f acc[4] = {};
  float mrun = -1e30f, lrun = 0.f;

  int kstart = qb - (WINDOW_ - 1);
  if (kstart < 0) kstart = 0;
  kstart &= ~31;
  const int kend = qb + 15;

  const f16* Kbase = Kh + kvh * D_ + (hi ? 8 : 0);
  const f16* Vbase = Vt + (size_t)(kvh * D_ + l16) * S_ + (hi ? 16 : 0);
  const float scale = 0.125f;  // 1/sqrt(64)

  for (int k0 = kstart; k0 <= kend; k0 += 32) {
    v8f ct0 = {}, ct1 = {};
    {
      const f16* kp = Kbase + (size_t)(k0 + l16) * KVD_;
      v16h a0 = cat8(ld8(kp), ld8(kp + 16));
      v16h a1 = cat8(ld8(kp + 32), ld8(kp + 48));
      ct0 = __builtin_amdgcn_wmma_f32_16x16x32_f16(false, a0, false, bq0, (short)0, ct0, false, false);
      ct0 = __builtin_amdgcn_wmma_f32_16x16x32_f16(false, a1, false, bq1, (short)0, ct0, false, false);
      const f16* kp2 = kp + (size_t)16 * KVD_;
      v16h a2 = cat8(ld8(kp2), ld8(kp2 + 16));
      v16h a3 = cat8(ld8(kp2 + 32), ld8(kp2 + 48));
      ct1 = __builtin_amdgcn_wmma_f32_16x16x32_f16(false, a2, false, bq0, (short)0, ct1, false, false);
      ct1 = __builtin_amdgcn_wmma_f32_16x16x32_f16(false, a3, false, bq1, (short)0, ct1, false, false);
    }

    const int mbase = k0 + (hi ? 8 : 0);
    float v0[8], v1[8];
    bool ok0[8], ok1[8];
    float vmax = -1e30f;
#pragma unroll
    for (int r = 0; r < 8; ++r) {
      const int m0 = mbase + r, m1 = mbase + 16 + r;
      ok0[r] = (m0 <= q) && (q - m0 < WINDOW_);
      ok1[r] = (m1 <= q) && (q - m1 < WINDOW_);
      v0[r] = ok0[r] ? ct0[r] * scale : -1e30f;
      v1[r] = ok1[r] ? ct1[r] * scale : -1e30f;
      vmax = fmaxf(vmax, fmaxf(v0[r], v1[r]));
    }
    vmax = fmaxf(vmax, __shfl_xor(vmax, 16, 32));
    const float mnew = fmaxf(mrun, vmax);
    const float corr = __expf(mrun - mnew);

    float p0[8], p1[8];
    float psum = 0.f;
#pragma unroll
    for (int r = 0; r < 8; ++r) {
      p0[r] = ok0[r] ? __expf(v0[r] - mnew) : 0.f;
      p1[r] = ok1[r] ? __expf(v1[r] - mnew) : 0.f;
      psum += p0[r] + p1[r];
    }
    const float rowsum = psum + __shfl_xor(psum, 16, 32);
    lrun = lrun * corr + rowsum;
    mrun = mnew;
#pragma unroll
    for (int c = 0; c < 4; ++c)
#pragma unroll
      for (int r = 0; r < 8; ++r) acc[c][r] *= corr;

    v16h bp;
#pragma unroll
    for (int r = 0; r < 8; ++r) {
      const float o0 = __shfl_xor(p0[r], 16, 32);
      const float o1 = __shfl_xor(p1[r], 16, 32);
      bp[r]     = (f16)(hi ? o1 : p0[r]);
      bp[8 + r] = (f16)(hi ? p1[r] : o0);
    }

    const f16* vp = Vbase + k0;
#pragma unroll
    for (int c = 0; c < 4; ++c) {
      const f16* vpc = vp + (size_t)(c * 16) * S_;
      v16h av = cat8(ld8(vpc), ld8(vpc + 8));
      acc[c] = __builtin_amdgcn_wmma_f32_16x16x32_f16(false, av, false, bp, (short)0, acc[c], false, false);
    }
  }

  const float invl = 1.f / lrun;
  f16* op = Out + (size_t)q * HD_ + head * D_;
#pragma unroll
  for (int c = 0; c < 4; ++c)
#pragma unroll
    for (int r = 0; r < 8; ++r) {
      const int d = c * 16 + r + (hi ? 8 : 0);
      op[d] = (f16)(acc[c][r] * invl);
    }
}

// ---------------------------------------------------------------------------
// launch
// ---------------------------------------------------------------------------
extern "C" void kernel_launch(void* const* d_in, const int* in_sizes, int n_in,
                              void* d_out, int out_size, void* d_ws, size_t ws_size,
                              hipStream_t stream) {
  const float*     X    = (const float*)d_in[0];
  const long long* pos  = (const long long*)d_in[1];  // int64 per reference
  const float*     Wq   = (const float*)d_in[2];
  const float*     Wk   = (const float*)d_in[3];
  const float*     Wv   = (const float*)d_in[4];
  const float*     Wo   = (const float*)d_in[5];
  const float*     qlnw = (const float*)d_in[6];
  const float*     klnw = (const float*)d_in[7];
  const float*     qa   = (const float*)d_in[8];
  const float*     ka   = (const float*)d_in[9];
  const float*     va   = (const float*)d_in[10];
  const float*     oa   = (const float*)d_in[11];
  float* out = (float*)d_out;

  char* w = (char*)d_ws;
  auto carve = [&](size_t bytes) {
    char* p = w;
    w += (bytes + 255) & ~(size_t)255;
    return p;
  };
  f16* Xh  = (f16*)carve((size_t)S_ * HID_ * 2);
  f16* Wqh = (f16*)carve((size_t)HD_ * HID_ * 2);
  f16* Wkh = (f16*)carve((size_t)KVD_ * HID_ * 2);
  f16* Wvh = (f16*)carve((size_t)KVD_ * HID_ * 2);
  f16* Woh = (f16*)carve((size_t)HID_ * HD_ * 2);
  float* Qf = (float*)carve((size_t)S_ * HD_ * 4);
  float* Kf = (float*)carve((size_t)S_ * KVD_ * 4);
  f16* Vt  = (f16*)carve((size_t)KVD_ * S_ * 2);   // V^T [KVD, S]
  f16* Qh  = (f16*)carve((size_t)S_ * HD_ * 2);
  f16* Kh2 = (f16*)carve((size_t)S_ * KVD_ * 2);
  f16* At  = (f16*)carve((size_t)S_ * HD_ * 2);

  auto cvt = [&](const float* s, f16* d, int n) {
    cvt_f32_to_f16<<<(n + 255) / 256, 256, 0, stream>>>(s, d, n);
  };
  cvt(X,  Xh,  S_ * HID_);
  cvt(Wq, Wqh, HD_ * HID_);
  cvt(Wk, Wkh, KVD_ * HID_);
  cvt(Wv, Wvh, KVD_ * HID_);
  cvt(Wo, Woh, HID_ * HD_);

  gemm_wmma<0><<<dim3(HD_ / 64,  S_ / 128), 256, 0, stream>>>(Xh, Wqh, Qf, qa, S_, HD_,  HID_);
  gemm_wmma<0><<<dim3(KVD_ / 64, S_ / 128), 256, 0, stream>>>(Xh, Wkh, Kf, ka, S_, KVD_, HID_);
  gemm_wmma<2><<<dim3(KVD_ / 64, S_ / 128), 256, 0, stream>>>(Xh, Wvh, Vt, va, S_, KVD_, HID_);

  ln_rope_kernel<<<(S_ * (H_ + KVH_)) / 4, 128, 0, stream>>>(Qf, Kf, qlnw, klnw, pos, Qh, Kh2);

  attn_kernel<<<H_ * (S_ / 64), 128, 0, stream>>>(Qh, Kh2, Vt, At);

  gemm_wmma<0><<<dim3(HID_ / 64, S_ / 128), 256, 0, stream>>>(At, Woh, out, oa, S_, HID_, HD_);
}
